// MultiHeadAttention_81973745811684
// MI455X (gfx1250) — matrix operand order repaired and checked by reading, not compiled, not run on hardware
//
#include <hip/hip_runtime.h>
#include <math.h>

// ---------------------------------------------------------------- constants
constexpr int Bk = 2;          // batch
constexpr int Sk = 2048;       // sequence
constexpr int Dk = 1024;       // embed dim
constexpr int Hk = 16;         // heads
constexpr int HDk = 64;        // head dim
constexpr int Mk = Bk * Sk;    // 4096 rows

// ---------------------------------------------------------------- WMMA types
typedef __attribute__((ext_vector_type(16))) __bf16 v16bf;
typedef __attribute__((ext_vector_type(8)))  float  v8f;

union FragAB { v16bf v; uint4 q[2]; };
union FragC  { v8f   v; float f[8]; };

// ---------------------------------------------------------------- async LDS
#if __has_builtin(__builtin_amdgcn_global_load_async_to_lds_b128)
#define HAVE_ASYNC_LDS 1
#else
#define HAVE_ASYNC_LDS 0
#pragma message("gfx1250: async-to-LDS builtin NOT available; using staged copies")
#endif

#if HAVE_ASYNC_LDS
typedef int v4i_vs __attribute__((vector_size(16)));
typedef __attribute__((address_space(1))) v4i_vs* gptr_b128;
typedef __attribute__((address_space(3))) v4i_vs* lptr_b128;

__device__ __forceinline__ void async_copy_b128(const unsigned short* g,
                                                unsigned short* l) {
  __builtin_amdgcn_global_load_async_to_lds_b128((gptr_b128)g, (lptr_b128)l, 0, 0);
}
__device__ __forceinline__ void wait_async0() {
#if __has_builtin(__builtin_amdgcn_s_wait_asynccnt)
  __builtin_amdgcn_s_wait_asynccnt(0);
#else
  asm volatile("s_wait_asynccnt 0" ::: "memory");
#endif
}
#endif

// ---------------------------------------------------------------- helpers
__device__ __forceinline__ unsigned short f2bf(float f) {
  unsigned int u = __float_as_uint(f);
  unsigned int r = u + 0x7fffu + ((u >> 16) & 1u);   // RNE
  return (unsigned short)(r >> 16);
}

__device__ __forceinline__ float fast_exp2(float x) {
#if __has_builtin(__builtin_amdgcn_exp2f)
  return __builtin_amdgcn_exp2f(x);
#else
  return exp2f(x);
#endif
}

// 16-lane butterfly reductions in pure VALU (DPP16):
// 0xB1=xor1, 0x4E=xor2, 0x141 row_half_mirror(=xor4 after quads uniform),
// 0x140 row_mirror(=xor8 after halves uniform).
__device__ __forceinline__ float dpp_max16(float v) {
  v = fmaxf(v, __int_as_float(__builtin_amdgcn_mov_dpp(__float_as_int(v), 0xB1, 0xF, 0xF, true)));
  v = fmaxf(v, __int_as_float(__builtin_amdgcn_mov_dpp(__float_as_int(v), 0x4E, 0xF, 0xF, true)));
  v = fmaxf(v, __int_as_float(__builtin_amdgcn_mov_dpp(__float_as_int(v), 0x141, 0xF, 0xF, true)));
  v = fmaxf(v, __int_as_float(__builtin_amdgcn_mov_dpp(__float_as_int(v), 0x140, 0xF, 0xF, true)));
  return v;
}
__device__ __forceinline__ float dpp_sum16(float v) {
  v += __int_as_float(__builtin_amdgcn_mov_dpp(__float_as_int(v), 0xB1, 0xF, 0xF, true));
  v += __int_as_float(__builtin_amdgcn_mov_dpp(__float_as_int(v), 0x4E, 0xF, 0xF, true));
  v += __int_as_float(__builtin_amdgcn_mov_dpp(__float_as_int(v), 0x141, 0xF, 0xF, true));
  v += __int_as_float(__builtin_amdgcn_mov_dpp(__float_as_int(v), 0x140, 0xF, 0xF, true));
  return v;
}

// A operand: 16(M) x 32(K) bf16 from row-major LDS; lane&15=row, lane>>4
// picks K offset 0/8; V0-3 = K base..base+7, V4-7 = K base+16..base+23.
__device__ __forceinline__ void load_frag_A(const unsigned short* lds, int ldr,
                                            int row0, int k0, int lane, FragAB& f) {
  const int h = lane >> 4, r = lane & 15;
  const unsigned short* p = lds + (size_t)(row0 + r) * ldr + k0 + 8 * h;
  f.q[0] = *(const uint4*)(p);
  f.q[1] = *(const uint4*)(p + 16);
}
// B operand: 32(K) x 16(N) bf16 from row-major LDS: lane&15 = N column,
// lane>>4 picks K offset 0/8; V0-3 = K base..base+7, V4-7 = K base+16..base+23.
__device__ __forceinline__ void load_frag_B(const unsigned short* lds, int ldr,
                                            int k0, int n0, int lane, FragAB& f) {
  const int h = lane >> 4, c = lane & 15;
  const unsigned short* p = lds + (size_t)(k0 + 8 * h) * ldr + n0 + c;
  union { uint4 q[2]; unsigned short u[16]; } t;
#pragma unroll
  for (int s = 0; s < 8; ++s) {
    t.u[s] = p[(size_t)s * ldr];
    t.u[s + 8] = p[(size_t)(s + 16) * ldr];
  }
  f.q[0] = t.q[0];
  f.q[1] = t.q[1];
}

// ---------------------------------------------------------------- convert
__global__ void f32_to_bf16_kernel(const float* __restrict__ in,
                                   unsigned short* __restrict__ out, int n4) {
  int i = blockIdx.x * blockDim.x + threadIdx.x;
  if (i < n4) {
    float4 f = ((const float4*)in)[i];
    ushort4 o;
    o.x = f2bf(f.x); o.y = f2bf(f.y); o.z = f2bf(f.z); o.w = f2bf(f.w);
    ((ushort4*)out)[i] = o;
  }
}

// ---------------------------------------------------------------- GEMM
// C[M,N] = X[M,K] @ W[K,N] + bias.  256 threads (8 waves), 128x128 tile,
// double-buffered LDS fed by async-to-LDS copies.  Wave (wm,wn) in a 4x2
// grid computes 32x64; B-fragments software-pipelined against WMMAs.
// MODE 0: bf16 scattered into [B,H,S,HD];  MODE 1: f32 [M,N].
template <int MODE>
__global__ __launch_bounds__(256)
void gemm_wmma_kernel(const unsigned short* __restrict__ X,
                      const unsigned short* __restrict__ W,
                      const float* __restrict__ bias,
                      void* __restrict__ out,
                      int M, int K, int N) {
  __shared__ unsigned short Xs[2][128][32 + 8];
  __shared__ unsigned short Ws[2][32][128 + 8];

  const int tid  = threadIdx.x;
  const int lane = tid & 31;
  const int wave = tid >> 5;
  const int wm = wave >> 1;          // 0..3  (M groups of 32)
  const int wn = wave & 1;           // 0..1  (N groups of 64)
  const int m0 = (int)blockIdx.x * 128;
  const int n0 = (int)blockIdx.y * 128;

  const int arow = tid >> 1, aseg = tid & 1;   // A tile 128x32
  const int brow = tid >> 3, bseg = tid & 7;   // B tile 32x128

#if HAVE_ASYNC_LDS
  auto issue_tile = [&](int kb, int buf) {
    const unsigned short* xs = X + (size_t)(m0 + arow) * K + kb + aseg * 16;
    unsigned short* xd = &Xs[buf][arow][aseg * 16];
    async_copy_b128(xs, xd);
    async_copy_b128(xs + 8, xd + 8);
    const unsigned short* wsrc = W + (size_t)(kb + brow) * N + n0 + bseg * 16;
    unsigned short* wd = &Ws[buf][brow][bseg * 16];
    async_copy_b128(wsrc, wd);
    async_copy_b128(wsrc + 8, wd + 8);
  };
#else
  uint4 ra0, ra1, rb0, rb1;
  auto load_regs = [&](int kb) {
    const uint4* as = (const uint4*)(X + (size_t)(m0 + arow) * K + kb + aseg * 16);
    ra0 = as[0]; ra1 = as[1];
    const uint4* bs = (const uint4*)(W + (size_t)(kb + brow) * N + n0 + bseg * 16);
    rb0 = bs[0]; rb1 = bs[1];
  };
  auto store_regs = [&](int buf) {
    uint4* ad = (uint4*)(&Xs[buf][arow][aseg * 16]);
    ad[0] = ra0; ad[1] = ra1;
    uint4* bd = (uint4*)(&Ws[buf][brow][bseg * 16]);
    bd[0] = rb0; bd[1] = rb1;
  };
#endif

  FragC acc[2][4];
#pragma unroll
  for (int mi = 0; mi < 2; ++mi)
#pragma unroll
    for (int ni = 0; ni < 4; ++ni)
#pragma unroll
      for (int i = 0; i < 8; ++i) acc[mi][ni].f[i] = 0.0f;

#if HAVE_ASYNC_LDS
  issue_tile(0, 0);
  wait_async0();
#else
  load_regs(0);
  store_regs(0);
#endif
  __syncthreads();

  int cur = 0;
  for (int k0 = 0; k0 < K; k0 += 32) {
    const bool has_next = (k0 + 32) < K;
    if (has_next) {
#if HAVE_ASYNC_LDS
      issue_tile(k0 + 32, cur ^ 1);
#else
      load_regs(k0 + 32);
#endif
    }

    FragAB a[2];
#pragma unroll
    for (int mi = 0; mi < 2; ++mi)
      load_frag_A(&Xs[cur][0][0], 32 + 8, wm * 32 + mi * 16, 0, lane, a[mi]);

    // software-pipeline B fragments against the WMMAs
    FragAB b_cur, b_nxt;
    load_frag_B(&Ws[cur][0][0], 128 + 8, 0, wn * 64, lane, b_cur);
#pragma unroll
    for (int ni = 0; ni < 4; ++ni) {
      if (ni < 3)
        load_frag_B(&Ws[cur][0][0], 128 + 8, 0, wn * 64 + (ni + 1) * 16, lane, b_nxt);
#pragma unroll
      for (int mi = 0; mi < 2; ++mi)
        acc[mi][ni].v = __builtin_amdgcn_wmma_f32_16x16x32_bf16(
            false, a[mi].v, false, b_cur.v, (short)0, acc[mi][ni].v, false, false);
      b_cur = b_nxt;
    }

    if (has_next) {
#if HAVE_ASYNC_LDS
      wait_async0();
#else
      store_regs(cur ^ 1);
#endif
    }
    __syncthreads();
    cur ^= 1;
  }

  const int col  = lane & 15;
  const int half = lane >> 4;
  if (MODE == 0) {
    unsigned short* o = (unsigned short*)out;
#pragma unroll
    for (int ni = 0; ni < 4; ++ni) {
      int n = n0 + wn * 64 + ni * 16 + col;
      int h = n >> 6, d = n & (HDk - 1);
      float bv = bias[n];
#pragma unroll
      for (int mi = 0; mi < 2; ++mi)
#pragma unroll
        for (int i = 0; i < 8; ++i) {
          int m = m0 + wm * 32 + mi * 16 + i + 8 * half;
          int b = m >> 11, s = m & (Sk - 1);
          o[((size_t)((b * Hk + h) * Sk + s)) * HDk + d] = f2bf(acc[mi][ni].f[i] + bv);
        }
    }
  } else {
    float* o = (float*)out;
#pragma unroll
    for (int ni = 0; ni < 4; ++ni) {
      int n = n0 + wn * 64 + ni * 16 + col;
      float bv = bias[n];
#pragma unroll
      for (int mi = 0; mi < 2; ++mi)
#pragma unroll
        for (int i = 0; i < 8; ++i) {
          int m = m0 + wm * 32 + mi * 16 + i + 8 * half;
          o[(size_t)m * N + n] = acc[mi][ni].f[i] + bv;
        }
    }
  }
}

// ---------------------------------------------------------------- attention
// One block = 64 query rows of one (b,h); 4 waves, 16 q-rows each.
// Flash attention, k-tiles of 64, double-buffered K/V, exp2-domain softmax,
// fragment loads software-pipelined against WMMAs.
__global__ __launch_bounds__(128)
void attn_wmma_kernel(const unsigned short* __restrict__ Qb,   // [B,H,S,HD]
                      const unsigned short* __restrict__ Kb,
                      const unsigned short* __restrict__ Vb,
                      unsigned short* __restrict__ ctx) {      // [B,S,D] bf16
  __shared__ unsigned short Qs[64][HDk + 8];
  __shared__ unsigned short KT[2][HDk][64 + 8];     // d x k (K transposed)
  __shared__ unsigned short Vs[2][64][HDk + 8];     // k x d
  __shared__ unsigned short Ps[4][16][64 + 8];      // per-wave P (q x k)

  const int tid  = threadIdx.x;
  const int lane = tid & 31;
  const int wave = tid >> 5;
  const int half = lane >> 4;
  const int col  = lane & 15;
  const int bh = (int)blockIdx.y;
  const int q0 = (int)blockIdx.x * 64;
  const size_t base = (size_t)bh * Sk * HDk;

  const int row = tid >> 1, seg = tid & 1;          // 64x64 tile loaders

  // K staging (transpose must go through registers)
  union Stage { uint4 q[4]; unsigned short u[32]; };
  Stage rk;
  auto load_k = [&](int kb) {
    const uint4* ks = (const uint4*)(Kb + base + (size_t)(kb + row) * HDk + seg * 32);
    rk.q[0] = ks[0]; rk.q[1] = ks[1]; rk.q[2] = ks[2]; rk.q[3] = ks[3];
  };
  auto store_k = [&](int buf) {
#pragma unroll
    for (int e = 0; e < 32; ++e) KT[buf][seg * 32 + e][row] = rk.u[e];
  };

#if HAVE_ASYNC_LDS
  auto issue_v = [&](int kb, int buf) {
    const unsigned short* vs = Vb + base + (size_t)(kb + row) * HDk + seg * 32;
    unsigned short* vd = &Vs[buf][row][seg * 32];
    async_copy_b128(vs, vd);
    async_copy_b128(vs + 8,  vd + 8);
    async_copy_b128(vs + 16, vd + 16);
    async_copy_b128(vs + 24, vd + 24);
  };
  {   // resident Q tile via async
    const unsigned short* qs = Qb + base + (size_t)(q0 + row) * HDk + seg * 32;
    unsigned short* qd = &Qs[row][seg * 32];
    async_copy_b128(qs, qd);
    async_copy_b128(qs + 8,  qd + 8);
    async_copy_b128(qs + 16, qd + 16);
    async_copy_b128(qs + 24, qd + 24);
  }
#else
  Stage rv;
  auto load_v = [&](int kb) {
    const uint4* vs = (const uint4*)(Vb + base + (size_t)(kb + row) * HDk + seg * 32);
    rv.q[0] = vs[0]; rv.q[1] = vs[1]; rv.q[2] = vs[2]; rv.q[3] = vs[3];
  };
  auto store_v = [&](int buf) {
    uint4* vd = (uint4*)(&Vs[buf][row][seg * 32]);
    vd[0] = rv.q[0]; vd[1] = rv.q[1]; vd[2] = rv.q[2]; vd[3] = rv.q[3];
  };
  {   // resident Q tile via direct copy
    const uint4* s = (const uint4*)(Qb + base + (size_t)(q0 + row) * HDk + seg * 32);
    uint4* d = (uint4*)(&Qs[row][seg * 32]);
    d[0] = s[0]; d[1] = s[1]; d[2] = s[2]; d[3] = s[3];
  }
#endif

  // ---- prologue: tile 0
  load_k(0);
  store_k(0);
#if HAVE_ASYNC_LDS
  issue_v(0, 0);
  wait_async0();
#else
  load_v(0);
  store_v(0);
#endif
  __syncthreads();

  FragAB qf[2];
  load_frag_A(&Qs[0][0], HDk + 8, wave * 16, 0,  lane, qf[0]);
  load_frag_A(&Qs[0][0], HDk + 8, wave * 16, 32, lane, qf[1]);

  FragC ctxacc[4];
#pragma unroll
  for (int t = 0; t < 4; ++t)
#pragma unroll
    for (int i = 0; i < 8; ++i) ctxacc[t].f[i] = 0.0f;

  float mrow[8], lrow[8];
#pragma unroll
  for (int i = 0; i < 8; ++i) { mrow[i] = -INFINITY; lrow[i] = 0.0f; }

  // scores scaled into exp2 domain: s' = s * (1/sqrt(64)) * log2(e)
  const float sscale = 0.125f * 1.44269504088896340736f;
  const int kend = q0 + 64;

  int cur = 0;
  for (int kk = 0; kk < kend; kk += 64) {
    const bool has_next = (kk + 64) < kend;
    if (has_next) {
      load_k(kk + 64);
#if HAVE_ASYNC_LDS
      issue_v(kk + 64, cur ^ 1);
#else
      load_v(kk + 64);
#endif
    }

    // ---- scores: four 16x16 subtiles, pipelined fragment loads
    FragC sc[4];
#pragma unroll
    for (int kt = 0; kt < 4; ++kt)
#pragma unroll
      for (int i = 0; i < 8; ++i) sc[kt].f[i] = 0.0f;

    {
      FragAB kb_cur[2], kb_nxt[2];
      load_frag_B(&KT[cur][0][0], 64 + 8, 0,  0, lane, kb_cur[0]);
      load_frag_B(&KT[cur][0][0], 64 + 8, 32, 0, lane, kb_cur[1]);
#pragma unroll
      for (int kt = 0; kt < 4; ++kt) {
        if (kt < 3) {
          load_frag_B(&KT[cur][0][0], 64 + 8, 0,  (kt + 1) * 16, lane, kb_nxt[0]);
          load_frag_B(&KT[cur][0][0], 64 + 8, 32, (kt + 1) * 16, lane, kb_nxt[1]);
        }
        sc[kt].v = __builtin_amdgcn_wmma_f32_16x16x32_bf16(
            false, qf[0].v, false, kb_cur[0].v, (short)0, sc[kt].v, false, false);
        sc[kt].v = __builtin_amdgcn_wmma_f32_16x16x32_bf16(
            false, qf[1].v, false, kb_cur[1].v, (short)0, sc[kt].v, false, false);
        kb_cur[0] = kb_nxt[0];
        kb_cur[1] = kb_nxt[1];
      }
    }

    // ---- scale + causal mask
#pragma unroll
    for (int kt = 0; kt < 4; ++kt) {
      int kg = kk + kt * 16 + col;
#pragma unroll
      for (int i = 0; i < 8; ++i) {
        int qg = q0 + wave * 16 + i + 8 * half;
        float v = sc[kt].f[i] * sscale;
        sc[kt].f[i] = (kg <= qg) ? v : -INFINITY;
      }
    }

    // ---- online softmax (exp2 domain), DPP reductions
    float alpha[8];
#pragma unroll
    for (int i = 0; i < 8; ++i) {
      float mx = fmaxf(fmaxf(sc[0].f[i], sc[1].f[i]), fmaxf(sc[2].f[i], sc[3].f[i]));
      mx = dpp_max16(mx);
      float mnew = fmaxf(mrow[i], mx);
      alpha[i] = fast_exp2(mrow[i] - mnew);
      mrow[i] = mnew;
      float rs = 0.0f;
#pragma unroll
      for (int kt = 0; kt < 4; ++kt) {
        float p = fast_exp2(sc[kt].f[i] - mnew);
        sc[kt].f[i] = p;
        rs += p;
      }
      rs = dpp_sum16(rs);
      lrow[i] = lrow[i] * alpha[i] + rs;
    }

#pragma unroll
    for (int t = 0; t < 4; ++t)
#pragma unroll
      for (int i = 0; i < 8; ++i) ctxacc[t].f[i] *= alpha[i];

    // ---- P -> per-wave LDS (C-layout scatter), reload as A operand
#pragma unroll
    for (int kt = 0; kt < 4; ++kt)
#pragma unroll
      for (int i = 0; i < 8; ++i)
        Ps[wave][i + 8 * half][kt * 16 + col] = f2bf(sc[kt].f[i]);
    __builtin_amdgcn_wave_barrier();
    asm volatile("s_wait_dscnt 0" ::: "memory");

    FragAB pf[2];
    load_frag_A(&Ps[wave][0][0], 64 + 8, 0, 0,  lane, pf[0]);
    load_frag_A(&Ps[wave][0][0], 64 + 8, 0, 32, lane, pf[1]);

    {   // ---- P @ V, pipelined fragment loads
      FragAB vb_cur[2], vb_nxt[2];
      load_frag_B(&Vs[cur][0][0], HDk + 8, 0,  0, lane, vb_cur[0]);
      load_frag_B(&Vs[cur][0][0], HDk + 8, 32, 0, lane, vb_cur[1]);
#pragma unroll
      for (int dt = 0; dt < 4; ++dt) {
        if (dt < 3) {
          load_frag_B(&Vs[cur][0][0], HDk + 8, 0,  (dt + 1) * 16, lane, vb_nxt[0]);
          load_frag_B(&Vs[cur][0][0], HDk + 8, 32, (dt + 1) * 16, lane, vb_nxt[1]);
        }
        ctxacc[dt].v = __builtin_amdgcn_wmma_f32_16x16x32_bf16(
            false, pf[0].v, false, vb_cur[0].v, (short)0, ctxacc[dt].v, false, false);
        ctxacc[dt].v = __builtin_amdgcn_wmma_f32_16x16x32_bf16(
            false, pf[1].v, false, vb_cur[1].v, (short)0, ctxacc[dt].v, false, false);
        vb_cur[0] = vb_nxt[0];
        vb_cur[1] = vb_nxt[1];
      }
    }

    if (has_next) {
      store_k(cur ^ 1);
#if HAVE_ASYNC_LDS
      wait_async0();
#else
      store_v(cur ^ 1);
#endif
    }
    __syncthreads();
    cur ^= 1;
  }

  // ---- normalize + store ctx bf16 [B,S,D]
  const int b = bh >> 4, h = bh & (Hk - 1);
#pragma unroll
  for (int dt = 0; dt < 4; ++dt) {
#pragma unroll
    for (int i = 0; i < 8; ++i) {
      int qrow = q0 + wave * 16 + i + 8 * half;
      float v = ctxacc[dt].f[i] / lrow[i];
      ctx[((size_t)(b * Sk + qrow)) * Dk + h * HDk + dt * 16 + col] = f2bf(v);
    }
  }
}

// ---------------------------------------------------------------- launch
extern "C" void kernel_launch(void* const* d_in, const int* in_sizes, int n_in,
                              void* d_out, int out_size, void* d_ws, size_t ws_size,
                              hipStream_t stream) {
  const float* x  = (const float*)d_in[0];
  const float* Wq = (const float*)d_in[1];
  const float* bq = (const float*)d_in[2];
  const float* Wk = (const float*)d_in[3];
  const float* bk = (const float*)d_in[4];
  const float* Wv = (const float*)d_in[5];
  const float* bv = (const float*)d_in[6];
  const float* Wo = (const float*)d_in[7];
  const float* bo = (const float*)d_in[8];

  char* ws = (char*)d_ws;
  size_t off = 0;
  auto carve = [&](size_t bytes) {
    void* p = ws + off;
    off += (bytes + 255) & ~(size_t)255;
    return p;
  };
  unsigned short* xb  = (unsigned short*)carve((size_t)Mk * Dk * 2);
  unsigned short* wqb = (unsigned short*)carve((size_t)Dk * Dk * 2);
  unsigned short* wkb = (unsigned short*)carve((size_t)Dk * Dk * 2);
  unsigned short* wvb = (unsigned short*)carve((size_t)Dk * Dk * 2);
  unsigned short* wob = (unsigned short*)carve((size_t)Dk * Dk * 2);
  unsigned short* Qb  = (unsigned short*)carve((size_t)Mk * Dk * 2);
  unsigned short* Kb  = (unsigned short*)carve((size_t)Mk * Dk * 2);
  unsigned short* Vb  = (unsigned short*)carve((size_t)Mk * Dk * 2);
  unsigned short* ctxb = xb;   // x dead after QKV projections

  const int nX4 = Mk * Dk / 4, nW4 = Dk * Dk / 4;
  f32_to_bf16_kernel<<<(nX4 + 255) / 256, 256, 0, stream>>>(x,  xb,  nX4);
  f32_to_bf16_kernel<<<(nW4 + 255) / 256, 256, 0, stream>>>(Wq, wqb, nW4);
  f32_to_bf16_kernel<<<(nW4 + 255) / 256, 256, 0, stream>>>(Wk, wkb, nW4);
  f32_to_bf16_kernel<<<(nW4 + 255) / 256, 256, 0, stream>>>(Wv, wvb, nW4);
  f32_to_bf16_kernel<<<(nW4 + 255) / 256, 256, 0, stream>>>(Wo, wob, nW4);

  dim3 gg(Mk / 128, Dk / 128);   // 32 x 8 blocks
  gemm_wmma_kernel<0><<<gg, 256, 0, stream>>>(xb, wqb, bq, Qb, Mk, Dk, Dk);
  gemm_wmma_kernel<0><<<gg, 256, 0, stream>>>(xb, wkb, bk, Kb, Mk, Dk, Dk);
  gemm_wmma_kernel<0><<<gg, 256, 0, stream>>>(xb, wvb, bv, Vb, Mk, Dk, Dk);

  dim3 ga(Sk / 64, Bk * Hk);     // 32 x 32 blocks
  attn_wmma_kernel<<<ga, 128, 0, stream>>>(Qb, Kb, Vb, ctxb);

  gemm_wmma_kernel<1><<<gg, 256, 0, stream>>>(ctxb, wob, bo, d_out, Mk, Dk, Dk);
}
